// ASIKT_47553877901675
// MI455X (gfx1250) — matrix operand order repaired and checked by reading, not compiled, not run on hardware
//
#include <hip/hip_runtime.h>

typedef _Float16 half_t;
typedef __attribute__((ext_vector_type(16))) _Float16 v16h;
typedef __attribute__((ext_vector_type(8)))  _Float16 v8h;
typedef __attribute__((ext_vector_type(8)))  float    v8f;
typedef unsigned int u32x4 __attribute__((ext_vector_type(4)));
typedef int          i32x4 __attribute__((ext_vector_type(4)));
typedef int          i32x8 __attribute__((ext_vector_type(8)));

#define SHUF16(lo, hi) __builtin_shufflevector(lo, hi, 0,1,2,3,4,5,6,7,8,9,10,11,12,13,14,15)

#if __has_builtin(__builtin_amdgcn_tensor_load_to_lds) && __has_builtin(__builtin_amdgcn_s_wait_tensorcnt)
#define USE_TDM 1
#else
#define USE_TDM 0
#endif

// ---------------- model dims ----------------
#define Bsz  32
#define Ssz  512
#define Dm   256
#define NHh  8
#define HD   32
#define Tt   16384        // Bsz*Ssz tokens
#define DFFc 2048

#if USE_TDM
// Build a 2D Tensor-DMA descriptor (D#) and issue TENSOR_LOAD_TO_LDS.
// Tile: tile_w x tile_h halves from a row-major [height x width] region
// (stride_elems halves per row), with optional LDS row padding
// (pad_interval/pad_amount codes per CDNA5 D# group1).
__device__ static inline void tdm_load_2d(const half_t* gaddr, unsigned lds_off,
                                          int width_elems, int height_rows,
                                          int tile_w, int tile_h, int stride_elems,
                                          int pad_interval, int pad_amount) {
  unsigned long long ga = (unsigned long long)gaddr;
  u32x4 g0;
  g0[0] = 1u;                                             // count=1, user descriptor
  g0[1] = lds_off;                                        // lds_addr (bytes)
  g0[2] = (unsigned)(ga & 0xFFFFFFFFu);                   // global_addr[31:0]
  g0[3] = (unsigned)((ga >> 32) & 0x01FFFFFFu) | (2u << 30); // addr[56:32] | type=2
  i32x8 g1;
  unsigned d0 = (1u << 16);                               // data_size = 2 bytes
  if (pad_amount) d0 |= (1u << 20) | ((unsigned)pad_interval << 22)
                      | ((unsigned)pad_amount << 25);     // pad_enable
  g1[0] = (int)d0;
  g1[1] = (int)(((unsigned)width_elems & 0xFFFFu) << 16); // tensor_dim0[15:0]
  g1[2] = (int)(((unsigned)width_elems >> 16) |
                (((unsigned)height_rows & 0xFFFFu) << 16)); // dim0 hi | dim1 lo
  g1[3] = (int)((((unsigned)height_rows >> 16) & 0xFFFFu) |
                ((unsigned)tile_w << 16));                // dim1 hi | tile_dim0
  g1[4] = (int)((unsigned)tile_h & 0xFFFFu);              // tile_dim1 | tile_dim2=0
  g1[5] = (int)(unsigned)stride_elems;                    // tensor_dim0_stride lo32
  g1[6] = 0;                                              // stride0 hi | stride1 lo
  g1[7] = 0;
  i32x4 gz4 = {0, 0, 0, 0};
  i32x8 gz8 = {0, 0, 0, 0, 0, 0, 0, 0};
  __builtin_amdgcn_tensor_load_to_lds(g0, g1, gz4, gz4, gz8, 0);
}
#endif

// ===================================================================
// f32 -> f16 convert (plus a column-padding variant for ragged weights)
// ===================================================================
__global__ void __launch_bounds__(256) k_cvt(const float* __restrict__ in,
                                             half_t* __restrict__ out, long n) {
  long i = (long)blockIdx.x * 256 + threadIdx.x;
  if (i < n) out[i] = (half_t)in[i];
}

__global__ void __launch_bounds__(256) k_cvt_pad(const float* __restrict__ in,
                                                 half_t* __restrict__ out,
                                                 int K, int N, int Np) {
  long i = (long)blockIdx.x * 256 + threadIdx.x;
  if (i >= (long)K * Np) return;
  int col = (int)(i % Np);
  long row = i / Np;
  out[i] = (col < N) ? (half_t)in[row * N + col] : (half_t)0.0f;
}

// ===================================================================
// Embedding prep: q_in, qa_in, kc_e, kca   (rasch branch of reference)
// ===================================================================
__global__ void __launch_bounds__(256) k_prep(
    const int* __restrict__ qd, const int* __restrict__ kd, const int* __restrict__ ad,
    const float* __restrict__ Ekc, const float* __restrict__ Ea,
    const float* __restrict__ Edev, const float* __restrict__ Ekcd, const float* __restrict__ Ead,
    float* __restrict__ q_in, float* __restrict__ qa_in,
    float* __restrict__ kc_e, float* __restrict__ kca) {
  long i = (long)blockIdx.x * 256 + threadIdx.x;
  if (i >= (long)Tt * Dm) return;
  long t = i >> 8; int d = (int)(i & 255);
  int qi = qd[t], ki = kd[t], ai = ad[t];
  float kce = Ekc[(size_t)ki * Dm + d];
  float ae  = Ea[(size_t)ai * Dm + d];
  float dev = Edev[qi];
  float kcd = Ekcd[(size_t)ki * Dm + d];
  float adv = Ead[(size_t)ai * Dm + d];
  kc_e[i] = kce;
  kca[i]  = kce + ae;
  q_in[i] = kce + dev * kcd;
  qa_in[i] = kce + ae + dev * (adv + kcd);
}

__device__ static inline float epi_act(float v, const float* bias, int n, int act) {
  v += bias ? bias[n] : 0.0f;
  if (act == 1) v = fmaxf(v, 0.0f);
  else if (act == 2) v = 1.0f / (1.0f + __expf(-v));
  return v;
}

// ===================================================================
// WMMA GEMM:  C[M,N] = act( A[M,K](f16) x B[K,N](f16) + bias )
// Requires M%64==0, K%32==0, N%128==0 (all call sites padded) -> no guards.
// Block tile 64(M) x 128(N); 8 waves, each a 2x2 grid of 16x16 WMMA tiles
// (4 WMMAs per 8 ds_load_b128 per K-step). A tile staged by the Tensor
// Data Mover (with LDS row padding); B tile staged transposed.
// ===================================================================
__global__ void __launch_bounds__(256) k_gemm(
    const half_t* __restrict__ A, const half_t* __restrict__ Bw,
    const float* __restrict__ bias, float* __restrict__ C,
    int M, int N, int K, int act) {
  __shared__ half_t As[64][40];    // 64 x 32 (+8 pad -> 80B row stride)
  __shared__ half_t Bt[128][40];   // B tile transposed: Bt[n][k]
  int tid = threadIdx.x;
  int lane = tid & 31, wave = tid >> 5;
  int bm = blockIdx.y * 64, bn = blockIdx.x * 128;
  int tr = wave >> 2, tc = wave & 3;     // 2x4 wave grid over 4x8 tile grid
  int tm0 = tr * 2, tm1 = tm0 + 1;
  int tn0 = tc * 2, tn1 = tn0 + 1;
  int l = lane & 15, hf = lane >> 4;
  v8f acc00 = {}, acc01 = {}, acc10 = {}, acc11 = {};
#if USE_TDM
  unsigned ldsA = (unsigned)(unsigned long long)&As[0][0];
#endif

  for (int k0 = 0; k0 < K; k0 += 32) {
#if USE_TDM
    if (tid == 0)
      tdm_load_2d(A + (size_t)bm * K + k0, ldsA, K - k0, M - bm, 32, 64, K,
                  /*pad_interval=*/3, /*pad_amount=*/3);  // 16-dword rows, +4-dword pad
#else
    { // manual A staging: one b128 per thread
      int row = tid >> 2, col = (tid & 3) * 8;
      *(int4*)&As[row][col] = *(const int4*)&A[(size_t)(bm + row) * K + k0 + col];
    }
#endif
    { // B staging: 32x128 transposed into Bt[n][k]; 16 halves per thread
      int row = tid >> 3, col = (tid & 7) * 16;
      const half_t* src = &Bw[(size_t)(k0 + row) * N + bn + col];
      union { int4 q; half_t h[8]; } u0, u1;
      u0.q = *(const int4*)src;
      u1.q = *(const int4*)(src + 8);
#pragma unroll
      for (int j = 0; j < 8; ++j) {
        Bt[col + j][row]     = u0.h[j];
        Bt[col + 8 + j][row] = u1.h[j];
      }
    }
#if USE_TDM
    __builtin_amdgcn_s_wait_tensorcnt(0);
#endif
    __syncthreads();
    if (k0 + 32 < K)   // gfx1250 global_prefetch_b8 for next B tile
      __builtin_prefetch(&Bw[(size_t)(k0 + 32 + (tid >> 3)) * N + bn + (tid & 7) * 16], 0, 1);

    // fragments: 2x ds_load_b128 each
    v16h a0 = SHUF16(*(const v8h*)&As[tm0 * 16 + l][hf * 8],
                     *(const v8h*)&As[tm0 * 16 + l][16 + hf * 8]);
    v16h a1 = SHUF16(*(const v8h*)&As[tm1 * 16 + l][hf * 8],
                     *(const v8h*)&As[tm1 * 16 + l][16 + hf * 8]);
    v16h b0 = SHUF16(*(const v8h*)&Bt[tn0 * 16 + l][hf * 8],
                     *(const v8h*)&Bt[tn0 * 16 + l][16 + hf * 8]);
    v16h b1 = SHUF16(*(const v8h*)&Bt[tn1 * 16 + l][hf * 8],
                     *(const v8h*)&Bt[tn1 * 16 + l][16 + hf * 8]);
    acc00 = __builtin_amdgcn_wmma_f32_16x16x32_f16(false, a0, false, b0, (short)0, acc00, false, false);
    acc01 = __builtin_amdgcn_wmma_f32_16x16x32_f16(false, a0, false, b1, (short)0, acc01, false, false);
    acc10 = __builtin_amdgcn_wmma_f32_16x16x32_f16(false, a1, false, b0, (short)0, acc10, false, false);
    acc11 = __builtin_amdgcn_wmma_f32_16x16x32_f16(false, a1, false, b1, (short)0, acc11, false, false);
    __syncthreads();
  }
#pragma unroll
  for (int r = 0; r < 8; ++r) {    // fully in-bounds: no guards
    int m0 = bm + tm0 * 16 + r + hf * 8;
    int m1 = bm + tm1 * 16 + r + hf * 8;
    int n0 = bn + tn0 * 16 + l;
    int n1 = bn + tn1 * 16 + l;
    C[(size_t)m0 * N + n0] = epi_act(acc00[r], bias, n0, act);
    C[(size_t)m0 * N + n1] = epi_act(acc01[r], bias, n1, act);
    C[(size_t)m1 * N + n0] = epi_act(acc10[r], bias, n0, act);
    C[(size_t)m1 * N + n1] = epi_act(acc11[r], bias, n1, act);
  }
}

// ===================================================================
// Flash attention, one wave per (b, head, 16-query tile).
// hd=32 -> QK^T is one WMMA per 16 keys; keys processed 32 at a time.
// Q/K fragments: two 16B runs per lane (halves 8h.., 16+8h..) -> global b128 loads.
// V staged transposed through LDS -> b128 fragment loads.
// ===================================================================
__global__ void __launch_bounds__(32) k_attn(
    const half_t* __restrict__ Qh, const half_t* __restrict__ Kh,
    const half_t* __restrict__ Vh, const float* __restrict__ dif,
    float* __restrict__ Out) {
  __shared__ half_t Ps[16][32];   // P tile (A-operand layout: row-major)
  __shared__ half_t Vt[32][40];   // V tile transposed: Vt[dh][key]
  int qt = blockIdx.x, h = blockIdx.y, b = blockIdx.z;
  int q0 = qt * 16;
  int lane = threadIdx.x;
  int l = lane & 15, hf = lane >> 4;
  const float scale = 0.17677669529663687f; // 1/sqrt(32)
  const size_t rs = (size_t)NHh * HD;       // 256

  const half_t* qrow = Qh + ((size_t)b * Ssz + q0 + l) * rs + h * HD;
  v16h aq = SHUF16(*(const v8h*)(qrow + hf * 8), *(const v8h*)(qrow + 16 + hf * 8));

  float mrow[8], lrow[8];
#pragma unroll
  for (int r = 0; r < 8; ++r) { mrow[r] = -1e30f; lrow[r] = 0.0f; }
  v8f o0 = {}, o1 = {};

  int kbmax = (q0 + 15) / 32;
  for (int kb = 0; kb <= kbmax; ++kb) {
    // ---- scores: two 16x16 WMMAs against K^T ----
    const half_t* krow0 = Kh + ((size_t)b * Ssz + (size_t)kb * 32 + l) * rs + h * HD;
    const half_t* krow1 = krow0 + 16 * rs;
    v16h bk0 = SHUF16(*(const v8h*)(krow0 + hf * 8), *(const v8h*)(krow0 + 16 + hf * 8));
    v16h bk1 = SHUF16(*(const v8h*)(krow1 + hf * 8), *(const v8h*)(krow1 + 16 + hf * 8));
    v8f z = {};
    v8f s0 = __builtin_amdgcn_wmma_f32_16x16x32_f16(false, aq, false, bk0, (short)0, z, false, false);
    v8f s1 = __builtin_amdgcn_wmma_f32_16x16x32_f16(false, aq, false, bk1, (short)0, z, false, false);

    // ---- V tile: each lane loads one key row, stores transposed ----
    const half_t* vrow = Vh + ((size_t)b * Ssz + (size_t)kb * 32 + lane) * rs + h * HD;
    v8h vr0 = *(const v8h*)(vrow);
    v8h vr1 = *(const v8h*)(vrow + 8);
    v8h vr2 = *(const v8h*)(vrow + 16);
    v8h vr3 = *(const v8h*)(vrow + 24);

    // ---- online softmax with dif-mask bias + causal mask ----
#pragma unroll
    for (int r = 0; r < 8; ++r) {
      int m = r + hf * 8;
      int qrow_i = q0 + m;
      int key0 = kb * 32 + l, key1 = key0 + 16;
      const float* drow = dif + ((size_t)b * Ssz + qrow_i) * Ssz;
      float v0 = s0[r] * scale + drow[key0];
      float v1 = s1[r] * scale + drow[key1];
      if (key0 > qrow_i) v0 = -1e9f;
      if (key1 > qrow_i) v1 = -1e9f;
      float mx = fmaxf(v0, v1);
#pragma unroll
      for (int off = 1; off < 16; off <<= 1) mx = fmaxf(mx, __shfl_xor(mx, off, 32));
      float mnew = fmaxf(mrow[r], mx);
      float alpha = __expf(mrow[r] - mnew);
      float p0 = __expf(v0 - mnew);
      float p1 = __expf(v1 - mnew);
      float ls = p0 + p1;
#pragma unroll
      for (int off = 1; off < 16; off <<= 1) ls += __shfl_xor(ls, off, 32);
      lrow[r] = lrow[r] * alpha + ls;
      mrow[r] = mnew;
      o0[r] *= alpha; o1[r] *= alpha;
      Ps[m][l]      = (half_t)p0;
      Ps[m][l + 16] = (half_t)p1;
    }
#pragma unroll
    for (int j = 0; j < 8; ++j) {
      Vt[j][lane]      = vr0[j];
      Vt[8 + j][lane]  = vr1[j];
      Vt[16 + j][lane] = vr2[j];
      Vt[24 + j][lane] = vr3[j];
    }
    __syncthreads();
    // ---- P x V: two 16x16 WMMAs accumulate into o ----
    v16h ap  = SHUF16(*(const v8h*)&Ps[l][hf * 8],       *(const v8h*)&Ps[l][16 + hf * 8]);
    v16h bv0 = SHUF16(*(const v8h*)&Vt[l][hf * 8],      *(const v8h*)&Vt[l][16 + hf * 8]);
    v16h bv1 = SHUF16(*(const v8h*)&Vt[l + 16][hf * 8], *(const v8h*)&Vt[l + 16][16 + hf * 8]);
    o0 = __builtin_amdgcn_wmma_f32_16x16x32_f16(false, ap, false, bv0, (short)0, o0, false, false);
    o1 = __builtin_amdgcn_wmma_f32_16x16x32_f16(false, ap, false, bv1, (short)0, o1, false, false);
    __syncthreads();
  }
  float* obase = Out + ((size_t)b * Ssz + q0) * rs + h * HD;
#pragma unroll
  for (int r = 0; r < 8; ++r) {
    int m = r + hf * 8;
    float inv = 1.0f / lrow[r];
    obase[(size_t)m * rs + l]      = o0[r] * inv;
    obase[(size_t)m * rs + l + 16] = o1[r] * inv;
  }
}

// ===================================================================
// LayerNorm over D=256 with residual add (one row / block)
// ===================================================================
__global__ void __launch_bounds__(256) k_ln(
    const float* __restrict__ x, const float* __restrict__ res,
    const float* __restrict__ w, const float* __restrict__ b,
    float* __restrict__ out) {
  __shared__ float red[256];
  long t = blockIdx.x;
  int d = threadIdx.x;
  float v = x[t * Dm + d] + (res ? res[t * Dm + d] : 0.0f);
  red[d] = v; __syncthreads();
  for (int off = 128; off; off >>= 1) { if (d < off) red[d] += red[d + off]; __syncthreads(); }
  float mean = red[0] / (float)Dm; __syncthreads();
  float c = v - mean;
  red[d] = c * c; __syncthreads();
  for (int off = 128; off; off >>= 1) { if (d < off) red[d] += red[d + off]; __syncthreads(); }
  float var = red[0] / (float)Dm;
  out[t * Dm + d] = c * rsqrtf(var + 1e-5f) * w[d] + b[d];
}

// ===================================================================
// Mamba: causal depthwise conv (width 4) + SiLU
// ===================================================================
__global__ void __launch_bounds__(256) k_conv(
    const float* __restrict__ zx, const float* __restrict__ cw,
    const float* __restrict__ cb, float* __restrict__ out,
    int Zn, int di, int convC) {
  long i = (long)blockIdx.x * 256 + threadIdx.x;
  long tot = (long)Tt * convC;
  if (i >= tot) return;
  int c = (int)(i % convC);
  long t = i / convC;
  int s = (int)(t & (Ssz - 1));
  float acc = cb[c];
#pragma unroll
  for (int k = 0; k < 4; ++k) {
    int ss = s - 3 + k;
    if (ss >= 0) acc += zx[(t - 3 + k) * Zn + di + c] * cw[c * 4 + k];
  }
  out[i] = acc / (1.0f + __expf(-acc));   // silu
}

// dt = softplus(dt_raw + bias); dA = exp(dt * (-exp(A_log)))
__global__ void __launch_bounds__(256) k_dt(
    const float* __restrict__ zx, const float* __restrict__ dtb,
    const float* __restrict__ Alog, float* __restrict__ dt_o,
    float* __restrict__ dA_o, int Zn, int off, int nh) {
  long i = (long)blockIdx.x * 256 + threadIdx.x;
  if (i >= (long)Tt * nh) return;
  int h = (int)(i % nh);
  long t = i / nh;
  float x = zx[t * Zn + off + h] + dtb[h];
  float dt = (x > 20.0f) ? x : __logf(1.0f + __expf(x));
  float A = -__expf(Alog[h]);
  dt_o[i] = dt;
  dA_o[i] = __expf(dt * A);
}

// ===================================================================
// Mamba selective scan: block = one (b, head); 64 threads = hd channels;
// state[ds=128] in registers; B/C broadcast via LDS per step.
// ===================================================================
__global__ void __launch_bounds__(64) k_scan(
    const float* __restrict__ xbc, const float* __restrict__ dt,
    const float* __restrict__ dA, const float* __restrict__ Dp,
    float* __restrict__ y, int nh, int di, int convC) {
  __shared__ float Bsh[128], Csh[128];
  int bh = blockIdx.x;
  int b = bh / nh, h = bh % nh;
  int dh = threadIdx.x;
  float st[128];
#pragma unroll
  for (int j = 0; j < 128; ++j) st[j] = 0.0f;
  float Dv = Dp[h];
  for (int s = 0; s < Ssz; ++s) {
    long t = (long)b * Ssz + s;
    const float* row = xbc + t * convC;
    __syncthreads();
#pragma unroll
    for (int j = 0; j < 2; ++j) {
      Bsh[dh + 64 * j] = row[di + dh + 64 * j];
      Csh[dh + 64 * j] = row[di + 128 + dh + 64 * j];
    }
    __syncthreads();
    float dAv = dA[t * nh + h];
    float dtv = dt[t * nh + h];
    float xv  = row[h * 64 + dh];
    float dtx = dtv * xv;
    float acc = 0.0f;
#pragma unroll
    for (int j = 0; j < 128; ++j) {
      st[j] = st[j] * dAv + dtx * Bsh[j];
      acc += st[j] * Csh[j];
    }
    y[t * di + h * 64 + dh] = acc + Dv * xv;
  }
}

// gated RMSNorm: out = (y * silu(z)) * norm_w * rsqrt(mean(v^2)+eps)
__global__ void __launch_bounds__(256) k_gate(
    const float* __restrict__ y, const float* __restrict__ zx,
    const float* __restrict__ nw, float* __restrict__ out, int di, int Zn) {
  __shared__ float red[256];
  long t = blockIdx.x;
  int tid = threadIdx.x;
  float ss = 0.0f;
  for (int c = tid; c < di; c += 256) {
    float z = zx[t * Zn + c];
    float g = z / (1.0f + __expf(-z));
    float v = y[t * di + c] * g;
    out[t * di + c] = v;
    ss += v * v;
  }
  red[tid] = ss; __syncthreads();
  for (int off = 128; off; off >>= 1) { if (tid < off) red[tid] += red[tid + off]; __syncthreads(); }
  float scale = rsqrtf(red[0] / (float)di + 1e-5f);
  for (int c = tid; c < di; c += 256) out[t * di + c] *= nw[c] * scale;
}

// xa = [qa_o | kca_o | shift1(q_o) | shift1(kc_o)]  (per-sequence shift)
__global__ void __launch_bounds__(256) k_build_xa(
    const float* __restrict__ qa_o, const float* __restrict__ kca_o,
    const float* __restrict__ q_o, const float* __restrict__ kc_o,
    float* __restrict__ xa) {
  long i = (long)blockIdx.x * 256 + threadIdx.x;
  if (i >= (long)Tt * 1024) return;
  long t = i >> 10; int c = (int)(i & 1023);
  int s = (int)(t & (Ssz - 1));
  float v;
  if (c < 256) v = qa_o[t * 256 + c];
  else if (c < 512) v = kca_o[t * 256 + (c - 256)];
  else {
    const float* src = (c < 768) ? q_o : kc_o;
    int cc = c & 255;
    v = (s < Ssz - 1) ? src[(t + 1) * 256 + cc] : 0.0f;
  }
  xa[i] = v;
}

__global__ void __launch_bounds__(256) k_concat2(
    const float* __restrict__ y, const float* __restrict__ kce, float* __restrict__ out) {
  long i = (long)blockIdx.x * 256 + threadIdx.x;
  if (i >= (long)Tt * 512) return;
  long t = i >> 9; int c = (int)(i & 511);
  out[i] = (c < 256) ? y[t * 256 + c] : kce[t * 256 + (c - 256)];
}

// final MLP layer (256 -> 1) as block dot product
__global__ void __launch_bounds__(256) k_dot(
    const float* __restrict__ x, const float* __restrict__ w,
    const float* __restrict__ b, float* __restrict__ out) {
  __shared__ float red[256];
  long t = blockIdx.x; int d = threadIdx.x;
  red[d] = x[t * 256 + d] * w[d]; __syncthreads();
  for (int off = 128; off; off >>= 1) { if (d < off) red[d] += red[d + off]; __syncthreads(); }
  if (d == 0) out[t] = red[0] + b[0];
}

// predictions + masked-BCE partial sums
__global__ void __launch_bounds__(256) k_final(
    const float* __restrict__ state, const float* __restrict__ slip,
    const float* __restrict__ guess, const int* __restrict__ target,
    float* __restrict__ preds_out, float* __restrict__ partial) {
  __shared__ float rb[256], rm[256];
  int tid = threadIdx.x;
  long i = (long)blockIdx.x * 256 + tid;
  float bce = 0.0f, msk = 0.0f;
  if (i < Tt) {
    float st = state[i], sl = slip[i], gu = guess[i];
    float p = (1.0f - st) * gu + st * (1.0f - sl);
    preds_out[i] = 1.0f / (1.0f + __expf(-p));
    float lab = (float)target[i];
    if (lab > -0.9f) {
      msk = 1.0f;
      bce = fmaxf(p, 0.0f) - p * lab + log1pf(__expf(-fabsf(p)));
    }
  }
  rb[tid] = bce; rm[tid] = msk; __syncthreads();
  for (int off = 128; off; off >>= 1) {
    if (tid < off) { rb[tid] += rb[tid + off]; rm[tid] += rm[tid + off]; }
    __syncthreads();
  }
  if (tid == 0) { partial[blockIdx.x * 2] = rb[0]; partial[blockIdx.x * 2 + 1] = rm[0]; }
}

__global__ void __launch_bounds__(64) k_reduce(
    const float* __restrict__ partial, float* __restrict__ out0,
    float* __restrict__ out1, int nb) {
  __shared__ float rb[64], rm[64];
  int t = threadIdx.x;
  rb[t] = (t < nb) ? partial[t * 2] : 0.0f;
  rm[t] = (t < nb) ? partial[t * 2 + 1] : 0.0f;
  __syncthreads();
  for (int off = 32; off; off >>= 1) {
    if (t < off) { rb[t] += rb[t + off]; rm[t] += rm[t + off]; }
    __syncthreads();
  }
  if (t == 0) { *out0 = rb[0]; *out1 = rm[0]; }
}

// ===================================================================
// Host-side orchestration
// ===================================================================
struct Arena { char* p; size_t off; };
static void* ar_alloc(Arena& a, size_t bytes) {
  size_t o = (a.off + 255) & ~(size_t)255;
  a.off = o + bytes;
  return a.p + o;
}
static float*  AF(Arena& a, long n) { return (float*)ar_alloc(a, (size_t)n * sizeof(float)); }
static half_t* AH(Arena& a, long n) { return (half_t*)ar_alloc(a, (size_t)n * sizeof(half_t)); }

static void cvt(hipStream_t st, const float* in, half_t* out, long n) {
  int g = (int)((n + 255) / 256);
  k_cvt<<<g, 256, 0, st>>>(in, out, n);
}
static void gemm(hipStream_t st, const half_t* A, const half_t* Bw, const float* bias,
                 float* C, int M, int N, int K, int act) {
  dim3 g(N / 128, M / 64);   // all shapes padded: N%128==0, M%64==0, K%32==0
  k_gemm<<<g, 256, 0, st>>>(A, Bw, bias, C, M, N, K, act);
}

struct StackW {
  half_t *wqkv, *wo, *w1, *w2;
  const float *bqkv, *bo, *ln1w, *ln1b, *ln2w, *ln2b, *b1, *b2;
};

static void run_stack(hipStream_t st, const StackW& W, const float* x_in,
                      const float* dif, float* x_out, Arena scratch) {
  long TD = (long)Tt * Dm;
  half_t* xh  = AH(scratch, TD);
  float*  qf  = AF(scratch, TD);
  float*  kf  = AF(scratch, TD);
  float*  vf  = AF(scratch, TD);
  half_t* qh  = AH(scratch, TD);
  half_t* kh  = AH(scratch, TD);
  half_t* vh  = AH(scratch, TD);
  float*  attn = AF(scratch, TD);
  half_t* ath = AH(scratch, TD);
  float*  proj = AF(scratch, TD);
  float*  xmid = AF(scratch, TD);
  half_t* xmh = AH(scratch, TD);
  float*  f1  = AF(scratch, (long)Tt * DFFc);
  half_t* f1h = AH(scratch, (long)Tt * DFFc);
  float*  f2  = AF(scratch, TD);

  const float* xcur = x_in;
  for (int i = 0; i < 2; ++i) {
    cvt(st, xcur, xh, TD);
    gemm(st, xh, W.wqkv + (size_t)(i * 3 + 0) * 65536, W.bqkv + (i * 3 + 0) * 256, qf, Tt, 256, 256, 0);
    gemm(st, xh, W.wqkv + (size_t)(i * 3 + 1) * 65536, W.bqkv + (i * 3 + 1) * 256, kf, Tt, 256, 256, 0);
    gemm(st, xh, W.wqkv + (size_t)(i * 3 + 2) * 65536, W.bqkv + (i * 3 + 2) * 256, vf, Tt, 256, 256, 0);
    cvt(st, qf, qh, TD); cvt(st, kf, kh, TD); cvt(st, vf, vh, TD);
    k_attn<<<dim3(Ssz / 16, NHh, Bsz), 32, 0, st>>>(qh, kh, vh, dif, attn);
    cvt(st, attn, ath, TD);
    gemm(st, ath, W.wo + (size_t)i * 65536, W.bo + i * 256, proj, Tt, 256, 256, 0);
    k_ln<<<Tt, 256, 0, st>>>(proj, xcur, W.ln1w + i * 256, W.ln1b + i * 256, xmid);
    cvt(st, xmid, xmh, TD);
    gemm(st, xmh, W.w1 + (size_t)i * 256 * 2048, W.b1 + i * 2048, f1, Tt, 2048, 256, 1);
    cvt(st, f1, f1h, (long)Tt * DFFc);
    gemm(st, f1h, W.w2 + (size_t)i * 2048 * 256, W.b2 + i * 256, f2, Tt, 256, 2048, 0);
    k_ln<<<Tt, 256, 0, st>>>(f2, xmid, W.ln2w + i * 256, W.ln2b + i * 256, x_out);
    xcur = x_out;
  }
}

struct MambaW {
  half_t *inp, *outp;
  const float *convw, *convb, *dtb, *Alog, *Dp, *nw;
  int dm, di, ds, nh, Znp;   // Znp = padded in_proj width (multiple of 128)
};

static void run_mamba(hipStream_t st, const MambaW& Wm, const float* x, float* out, Arena scratch) {
  int Znp = Wm.Znp;                       // padded row stride of zxbcdt
  int convC = Wm.di + 2 * Wm.ds;
  half_t* xh   = AH(scratch, (long)Tt * Wm.dm);
  float*  zx   = AF(scratch, (long)Tt * Znp);
  float*  xbc  = AF(scratch, (long)Tt * convC);
  float*  dt_  = AF(scratch, (long)Tt * Wm.nh);
  float*  dA_  = AF(scratch, (long)Tt * Wm.nh);
  float*  ysc  = AF(scratch, (long)Tt * Wm.di);
  float*  gat  = AF(scratch, (long)Tt * Wm.di);
  half_t* gath = AH(scratch, (long)Tt * Wm.di);

  cvt(st, x, xh, (long)Tt * Wm.dm);
  gemm(st, xh, Wm.inp, nullptr, zx, Tt, Znp, Wm.dm, 0);
  long ctot = (long)Tt * convC;
  k_conv<<<(int)((ctot + 255) / 256), 256, 0, st>>>(zx, Wm.convw, Wm.convb, xbc, Znp, Wm.di, convC);
  long ntot = (long)Tt * Wm.nh;
  k_dt<<<(int)((ntot + 255) / 256), 256, 0, st>>>(zx, Wm.dtb, Wm.Alog, dt_, dA_, Znp, 2 * Wm.di + 2 * Wm.ds, Wm.nh);
  k_scan<<<Bsz * Wm.nh, 64, 0, st>>>(xbc, dt_, dA_, Wm.Dp, ysc, Wm.nh, Wm.di, convC);
  k_gate<<<Tt, 256, 0, st>>>(ysc, zx, Wm.nw, gat, Wm.di, Znp);
  cvt(st, gat, gath, (long)Tt * Wm.di);
  gemm(st, gath, Wm.outp, nullptr, out, Tt, Wm.dm, Wm.di, 0);
}

static void run_head(hipStream_t st, const half_t* w1h, const float* b1,
                     const half_t* w2h, const float* b2,
                     const float* w3, const float* b3,
                     const half_t* ch, float* outv, Arena scratch) {
  float*  h1  = AF(scratch, (long)Tt * 512);
  half_t* h1h = AH(scratch, (long)Tt * 512);
  float*  h2  = AF(scratch, (long)Tt * 256);
  gemm(st, ch, w1h, b1, h1, Tt, 512, 512, 1);
  cvt(st, h1, h1h, (long)Tt * 512);
  gemm(st, h1h, w2h, b2, h2, Tt, 256, 512, 1);
  k_dot<<<Tt, 256, 0, st>>>(h2, w3, b3, outv);
}

extern "C" void kernel_launch(void* const* d_in, const int* in_sizes, int n_in,
                              void* d_out, int out_size, void* d_ws, size_t ws_size,
                              hipStream_t stream) {
  (void)in_sizes; (void)n_in; (void)out_size; (void)ws_size;
  hipStream_t st = stream;

  // --- input pointers (setup_inputs dict insertion order) ---
  const int*   q_data  = (const int*)d_in[0];
  const int*   kc_data = (const int*)d_in[1];
  const int*   a_data  = (const int*)d_in[2];
  const int*   target  = (const int*)d_in[3];
  const float* q_dif   = (const float*)d_in[4];
  const float* qa_dif  = (const float*)d_in[5];
  const float* Ekc  = (const float*)d_in[7];
  const float* Ea   = (const float*)d_in[8];
  const float* Edev = (const float*)d_in[9];
  const float* Ekcd = (const float*)d_in[10];
  const float* Ead  = (const float*)d_in[11];

  Arena A{(char*)d_ws, 0};

  // --- convert weights to f16 once per call (deterministic) ---
  StackW SW[4];
  for (int sI = 0; sI < 4; ++sI) {
    int tb = 12 + sI * 12;
    SW[sI].wqkv = AH(A, 393216);  cvt(st, (const float*)d_in[tb + 0], SW[sI].wqkv, 393216);
    SW[sI].wo   = AH(A, 131072);  cvt(st, (const float*)d_in[tb + 2], SW[sI].wo,   131072);
    SW[sI].w1   = AH(A, 1048576); cvt(st, (const float*)d_in[tb + 8], SW[sI].w1,   1048576);
    SW[sI].w2   = AH(A, 1048576); cvt(st, (const float*)d_in[tb + 10], SW[sI].w2,  1048576);
    SW[sI].bqkv = (const float*)d_in[tb + 1];
    SW[sI].bo   = (const float*)d_in[tb + 3];
    SW[sI].ln1w = (const float*)d_in[tb + 4];
    SW[sI].ln1b = (const float*)d_in[tb + 5];
    SW[sI].ln2w = (const float*)d_in[tb + 6];
    SW[sI].ln2b = (const float*)d_in[tb + 7];
    SW[sI].b1   = (const float*)d_in[tb + 9];
    SW[sI].b2   = (const float*)d_in[tb + 11];
  }
  MambaW M5, M6;
  M5.dm = 1024; M5.di = 2048; M5.ds = 128; M5.nh = 32; M5.Znp = 4480;  // 4384 -> 4480
  M5.inp  = AH(A, (long)1024 * 4480);
  { long n = (long)1024 * 4480;
    k_cvt_pad<<<(int)((n + 255) / 256), 256, 0, st>>>((const float*)d_in[60], M5.inp, 1024, 4384, 4480); }
  M5.outp = AH(A, 2097152); cvt(st, (const float*)d_in[67], M5.outp, 2097152);
  M5.convw = (const float*)d_in[61]; M5.convb = (const float*)d_in[62];
  M5.dtb = (const float*)d_in[63];   M5.Alog = (const float*)d_in[64];
  M5.Dp = (const float*)d_in[65];    M5.nw = (const float*)d_in[66];
  M6.dm = 256; M6.di = 512; M6.ds = 128; M6.nh = 8; M6.Znp = 1408;     // 1288 -> 1408
  M6.inp  = AH(A, (long)256 * 1408);
  { long n = (long)256 * 1408;
    k_cvt_pad<<<(int)((n + 255) / 256), 256, 0, st>>>((const float*)d_in[68], M6.inp, 256, 1288, 1408); }
  M6.outp = AH(A, 131072);  cvt(st, (const float*)d_in[75], M6.outp, 131072);
  M6.convw = (const float*)d_in[69]; M6.convb = (const float*)d_in[70];
  M6.dtb = (const float*)d_in[71];   M6.Alog = (const float*)d_in[72];
  M6.Dp = (const float*)d_in[73];    M6.nw = (const float*)d_in[74];
  half_t* fuse_wh = AH(A, 262144); cvt(st, (const float*)d_in[76], fuse_wh, 262144);
  const float* fuse_b = (const float*)d_in[77];
  half_t* hw1[3]; half_t* hw2[3];
  const float *hb1[3], *hb2[3], *hw3[3], *hb3[3];
  for (int hI = 0; hI < 3; ++hI) {
    int hb = 78 + hI * 6;
    hw1[hI] = AH(A, 262144); cvt(st, (const float*)d_in[hb + 0], hw1[hI], 262144);
    hw2[hI] = AH(A, 131072); cvt(st, (const float*)d_in[hb + 2], hw2[hI], 131072);
    hb1[hI] = (const float*)d_in[hb + 1];
    hb2[hI] = (const float*)d_in[hb + 3];
    hw3[hI] = (const float*)d_in[hb + 4];
    hb3[hI] = (const float*)d_in[hb + 5];
  }

  // --- persistent activations ---
  long TD = (long)Tt * Dm;
  float* kc_e  = AF(A, TD);
  float* q_in  = AF(A, TD);
  float* qa_in = AF(A, TD);
  float* kca   = AF(A, TD);
  float* q_o   = AF(A, TD);
  float* qa_o  = AF(A, TD);
  float* kc_o  = AF(A, TD);
  float* kca_o = AF(A, TD);
  float* xa    = AF(A, (long)Tt * 1024);
  float* h5    = AF(A, (long)Tt * 1024);
  half_t* h5h  = AH(A, (long)Tt * 1024);
  float* fuseo = AF(A, TD);
  float* y6    = AF(A, TD);
  float* conc  = AF(A, (long)Tt * 512);
  half_t* conch = AH(A, (long)Tt * 512);
  float* stv = AF(A, Tt);
  float* slv = AF(A, Tt);
  float* guv = AF(A, Tt);
  float* partial = AF(A, 256);

  // --- forward pass ---
  k_prep<<<(int)((TD + 255) / 256), 256, 0, st>>>(q_data, kc_data, a_data,
      Ekc, Ea, Edev, Ekcd, Ead, q_in, qa_in, kc_e, kca);

  run_stack(st, SW[0], q_in,  q_dif,  q_o,   A);   // Arena copied: scratch auto-reset
  run_stack(st, SW[1], qa_in, qa_dif, qa_o,  A);
  run_stack(st, SW[2], kc_e,  q_dif,  kc_o,  A);
  run_stack(st, SW[3], kca,   qa_dif, kca_o, A);

  long xtot = (long)Tt * 1024;
  k_build_xa<<<(int)((xtot + 255) / 256), 256, 0, st>>>(qa_o, kca_o, q_o, kc_o, xa);

  run_mamba(st, M5, xa, h5, A);
  cvt(st, h5, h5h, (long)Tt * 1024);
  gemm(st, h5h, fuse_wh, fuse_b, fuseo, Tt, 256, 1024, 2);  // sigmoid
  run_mamba(st, M6, fuseo, y6, A);

  long ctot = (long)Tt * 512;
  k_concat2<<<(int)((ctot + 255) / 256), 256, 0, st>>>(y6, kc_e, conc);
  cvt(st, conc, conch, ctot);

  run_head(st, hw1[0], hb1[0], hw2[0], hb2[0], hw3[0], hb3[0], conch, stv, A);
  run_head(st, hw1[1], hb1[1], hw2[1], hb2[1], hw3[1], hb3[1], conch, slv, A);
  run_head(st, hw1[2], hb1[2], hw2[2], hb2[2], hw3[2], hb3[2], conch, guv, A);

  float* out_f = (float*)d_out;
  k_final<<<(Tt + 255) / 256, 256, 0, st>>>(stv, slv, guv, target, out_f + 1, partial);
  k_reduce<<<1, 64, 0, st>>>(partial, out_f, out_f + 1 + Tt, (Tt + 255) / 256);
}
